// ptBEVnet_26379689132549
// MI455X (gfx1250) — matrix operand order repaired and checked by reading, not compiled, not run on hardware
//
#include <hip/hip_runtime.h>
#include <stdint.h>

typedef __attribute__((ext_vector_type(16))) _Float16 v16h;
typedef __attribute__((ext_vector_type(8)))  float    v8f;

#define GXd 480
#define GYd 360
#define NHd 32
#define HWd (GXd*GYd)          // 172800
#define Bd  2
#define Nd  65536
#define FEAd 9
#define C1d 64
#define C2d 128
#define C3d 256
#define C4d 512
#define MAXPTd 256u
#define ENC_NEG_INF 0x007FFFFFu   // order-preserving encoding of -inf

__device__ __forceinline__ unsigned encf(float f) {
  unsigned u = __float_as_uint(f);
  return (u & 0x80000000u) ? ~u : (u | 0x80000000u);
}
__device__ __forceinline__ float decf(unsigned e) {
  return (e & 0x80000000u) ? __uint_as_float(e & 0x7FFFFFFFu)
                           : __uint_as_float(~e);
}

// A fragment: 16x32 f16, row-major LDS tile (pitch halves), k-chunk base kb.
// ISA layout: lane L -> row L%16; halves h: K = h + 8*((h>=8)+(L>=16)).
__device__ __forceinline__ v16h load_A(const _Float16* tile, int pitch, int kb, int lane) {
  int row  = lane & 15;
  int koff = (lane >> 4) << 3;          // 0 or 8
  const _Float16* rp = tile + row * pitch + kb;
  v16h a;
#pragma unroll
  for (int r = 0; r < 8; ++r) {
    int k0 = 2 * r + ((r >= 4) ? 8 : 0) + koff;
    a[2 * r]     = rp[k0];
    a[2 * r + 1] = rp[k0 + 1];
  }
  return a;
}

// B fragment: 32x16 f16 from transposed LDS weights wt[n][pitch], k-chunk kb.
// Layout: lane L -> col L%16; VGPR r: K = 2r,2r+1 (+8 for r>=4, +8 for lanes>=16).
__device__ __forceinline__ v16h load_B(const _Float16* wt, int pitch, int kb, int lane) {
  int n    = lane & 15;
  int koff = (lane >> 4) << 3;          // 0 or 8
  const _Float16* np = wt + n * pitch + kb + koff;
  v16h b;
#pragma unroll
  for (int r = 0; r < 8; ++r) {
    int k0 = 2 * r + ((r >= 4) ? 8 : 0);
    b[2 * r]     = np[k0];
    b[2 * r + 1] = np[k0 + 1];
  }
  return b;
}

__device__ __forceinline__ v8f wmma_f16(v16h a, v16h b, v8f c) {
  return __builtin_amdgcn_wmma_f32_16x16x32_f16(false, a, false, b, (short)0, c, false, false);
}

// One MLP layer for this wave's 16-row tile: xout = act((xin @ W + b)*s + t).
// Column tiles processed in groups of 4 so each A fragment feeds 4 WMMAs.
__device__ __forceinline__ void gemm_layer(const _Float16* xin, int kp,
                                           const _Float16* wt,     // [cout][kp] transposed
                                           const float* S, const float* T, const float* Bv,
                                           _Float16* xout, int cout,
                                           int ksteps, int lane) {
  for (int cg = 0; cg < (cout >> 6); ++cg) {       // groups of 4 column tiles
    v8f acc[4] = {{}, {}, {}, {}};
    for (int ks = 0; ks < ksteps; ++ks) {
      v16h a = load_A(xin, kp, ks * 32, lane);
#pragma unroll
      for (int c = 0; c < 4; ++c) {
        v16h b = load_B(wt + (cg * 4 + c) * 16 * kp, kp, ks * 32, lane);
        acc[c] = wmma_f16(a, b, acc[c]);
      }
    }
    int n  = lane & 15;
    int mo = (lane >> 4) << 3;
#pragma unroll
    for (int c = 0; c < 4; ++c) {
      int col = (cg * 4 + c) * 16 + n;
      float sc = S[col], sh = T[col], bb = Bv[col];
#pragma unroll
      for (int r = 0; r < 8; ++r) {
        float v = fmaxf((acc[c][r] + bb) * sc + sh, 0.0f);
        xout[(r + mo) * cout + col] = (_Float16)v;
      }
    }
  }
}

// ---------------------------------------------------------------- zero counters
__global__ void zero_kernel(unsigned* __restrict__ cnt, unsigned* __restrict__ cnt2) {
  int i = blockIdx.x * blockDim.x + threadIdx.x;
  int stride = gridDim.x * blockDim.x;
  for (int j = i; j < HWd; j += stride) { cnt[j] = 0u; cnt2[j] = 0u; }
}

// ---------------------------------------------------------------- count points per voxel
__global__ void count_kernel(const int* __restrict__ xy, unsigned* __restrict__ cnt, int batch) {
  int p = blockIdx.x * blockDim.x + threadIdx.x;
  if (p < Nd) {
    int gx = xy[((long)(batch * Nd + p) << 1) + 0];
    int gy = xy[((long)(batch * Nd + p) << 1) + 1];
    atomicAdd(&cnt[gx * GYd + gy], 1u);
  }
}

// ---------------------------------------------------------------- init enc rows (occupied voxels only)
__global__ void init_kernel(unsigned* __restrict__ enc, const unsigned* __restrict__ cnt) {
  long i = (long)blockIdx.x * blockDim.x + threadIdx.x;
  long stride = (long)gridDim.x * blockDim.x;
  long total = (long)HWd * (C4d / 4);               // uint4 groups
  uint4 val = make_uint4(ENC_NEG_INF, ENC_NEG_INF, ENC_NEG_INF, ENC_NEG_INF);
  for (long j = i; j < total; j += stride) {
    int v = (int)(j >> 7);                          // 128 uint4 per voxel row
    if (cnt[v] > 0u) ((uint4*)enc)[j] = val;
  }
}

// ---------------------------------------------------------------- fused MLP + scatter max
__global__ __launch_bounds__(256)
void mlp_scatter_kernel(const float* __restrict__ pt_fea, const int* __restrict__ xy,
                        const float* __restrict__ s0, const float* __restrict__ t0,
                        const float* __restrict__ s1, const float* __restrict__ t1,
                        const float* __restrict__ s2, const float* __restrict__ t2,
                        const float* __restrict__ s3, const float* __restrict__ t3,
                        const float* __restrict__ W1, const float* __restrict__ b1,
                        const float* __restrict__ W2, const float* __restrict__ b2,
                        const float* __restrict__ W3, const float* __restrict__ b3,
                        const float* __restrict__ W4, const float* __restrict__ b4,
                        unsigned* __restrict__ enc, unsigned* __restrict__ cnt2, int batch) {
  extern __shared__ char smem[];
  _Float16* w1t = (_Float16*)smem;           // [64][32]
  _Float16* w2t = w1t + 64 * 32;             // [128][64]
  _Float16* w3t = w2t + 128 * 64;            // [256][128]
  _Float16* w4c = w3t + 256 * 128;           // [16][256] streamed chunk
  _Float16* xbb = w4c + 16 * 256;            // 8 waves * 2 * 16*256
  float* par = (float*)(xbb + 8 * 2 * 16 * 256);
  float* S1 = par;        float* T1 = S1 + 64;  float* B1 = T1 + 64;
  float* S2 = B1 + 64;    float* T2 = S2 + 128; float* B2 = T2 + 128;
  float* S3 = B2 + 128;   float* T3 = S3 + 256; float* B3 = T3 + 256;
  float* B4 = B3 + 256;
  int* lin  = (int*)(B4 + 512);              // [128]
  int* keep = lin + 128;                     // [128]

  int t = threadIdx.x, lane = t & 31, wave = t >> 5;
  int base = blockIdx.x * 128;

  // stage transposed weights (coalesced global reads)
  for (int i = t; i < 64 * 32; i += 256)   { int k = i >> 6, n = i & 63;  w1t[n * 32 + k]  = (_Float16)((k < FEAd) ? W1[k * C1d + n] : 0.0f); }
  for (int i = t; i < 128 * 64; i += 256)  { int k = i >> 7, n = i & 127; w2t[n * 64 + k]  = (_Float16)W2[k * C2d + n]; }
  for (int i = t; i < 256 * 128; i += 256) { int k = i >> 8, n = i & 255; w3t[n * 128 + k] = (_Float16)W3[k * C3d + n]; }
  for (int i = t; i < 64;  i += 256) { S1[i] = s1[i]; T1[i] = t1[i]; B1[i] = b1[i]; }
  for (int i = t; i < 128; i += 256) { S2[i] = s2[i]; T2[i] = t2[i]; B2[i] = b2[i]; }
  for (int i = t; i < 256; i += 256) { S3[i] = s3[i]; T3[i] = t3[i]; B3[i] = b3[i]; }
  for (int i = t; i < 512; i += 256) { B4[i] = b4[i]; }

  // voxel id + arrival rank (keep-cap)
  if (t < 128) {
    int p  = base + t;
    int gx = xy[((long)(batch * Nd + p) << 1) + 0];
    int gy = xy[((long)(batch * Nd + p) << 1) + 1];
    int l  = gx * GYd + gy;
    lin[t] = l;
    unsigned r = atomicAdd(&cnt2[l], 1u);
    keep[t] = (r < MAXPTd) ? 1 : 0;
  }

  // stage x0 (16x32, affine s0/t0, zero-padded K) into this wave's buffer A
  _Float16* xa = xbb + wave * (2 * 16 * 256);
  _Float16* xc = xa + 16 * 256;
  for (int i = lane; i < 16 * 32; i += 32) {
    int row = i >> 5, k = i & 31;
    int p = base + wave * 16 + row;
    float v = 0.0f;
    if (k < FEAd) v = pt_fea[(long)(batch * Nd + p) * FEAd + k] * s0[k] + t0[k];
    xa[row * 32 + k] = (_Float16)v;
  }
  __syncthreads();

  // layers 1..3: per-wave, weights are read-only in LDS
  gemm_layer(xa, 32,  w1t, S1, T1, B1, xc, C1d, 1, lane);   // x1 -> xc[16][64]
  gemm_layer(xc, 64,  w2t, S2, T2, B2, xa, C2d, 2, lane);   // x2 -> xa[16][128]
  gemm_layer(xa, 128, w3t, S3, T3, B3, xc, C3d, 4, lane);   // x3 -> xc[16][256]

  // layer 4: hoist this wave's 8 A fragments (x3 is stable), stream W4 chunks
  v16h a4[8];
#pragma unroll
  for (int ks = 0; ks < 8; ++ks) a4[ks] = load_A(xc, 256, ks * 32, lane);

  for (int ct = 0; ct < 32; ++ct) {
    __syncthreads();
    for (int i = t; i < 16 * 256; i += 256) {
      int k = i >> 4, n = i & 15;
      w4c[n * 256 + k] = (_Float16)W4[k * C4d + ct * 16 + n];
    }
    __syncthreads();
    v8f acc = {};
#pragma unroll
    for (int ks = 0; ks < 8; ++ks) {
      v16h b = load_B(w4c, 256, ks * 32, lane);
      acc = wmma_f16(a4[ks], b, acc);
    }
    int n  = lane & 15;
    int mo = (lane >> 4) << 3;
    int col = ct * 16 + n;
    float bb = B4[col];
#pragma unroll
    for (int r = 0; r < 8; ++r) {
      int pl = wave * 16 + r + mo;
      if (keep[pl]) {
        float v = acc[r] + bb;
        atomicMax(&enc[(long)lin[pl] * C4d + col], encf(v));
      }
    }
  }
}

// ---------------------------------------------------------------- compression 512->32 + transpose
__global__ __launch_bounds__(256)
void compress_kernel(const unsigned* __restrict__ enc, const unsigned* __restrict__ cnt,
                     const float* __restrict__ Wc, const float* __restrict__ bc,
                     float* __restrict__ out, int batch) {
  extern __shared__ char smem[];
  _Float16* wct = (_Float16*)smem;          // [32][512] transposed
  _Float16* xt  = wct + 32 * 512;           // 8 waves * 16*512
  float* BC = (float*)(xt + 8 * 16 * 512);  // [32]

  int t = threadIdx.x, lane = t & 31, wave = t >> 5;
  for (int i = t; i < 32 * 512; i += 256) {
    int k = i >> 5, n = i & 31;
    wct[n * 512 + k] = (_Float16)Wc[k * NHd + n];
  }
  if (t < NHd) BC[t] = bc[t];

  int vbase = blockIdx.x * 128;
  _Float16* xw = xt + wave * 16 * 512;
  for (int i = lane; i < 16 * 512; i += 32) {
    int row = i >> 9, k = i & 511;
    int v = vbase + wave * 16 + row;
    float f = 0.0f;
    if (cnt[v] > 0u) f = decf(enc[(long)v * C4d + k]);   // skip HBM read for empty voxels
    xw[i] = (_Float16)f;
  }
  __syncthreads();

  v8f acc0 = {}, acc1 = {};
#pragma unroll 4
  for (int ks = 0; ks < 16; ++ks) {
    v16h a  = load_A(xw, 512, ks * 32, lane);
    acc0 = wmma_f16(a, load_B(wct,            512, ks * 32, lane), acc0);
    acc1 = wmma_f16(a, load_B(wct + 16 * 512, 512, ks * 32, lane), acc1);
  }

  int n  = lane & 15;
  int mo = (lane >> 4) << 3;
#pragma unroll
  for (int c = 0; c < 2; ++c) {
    int ch = c * 16 + n;
    float bb = BC[ch];
    const v8f& acc = c ? acc1 : acc0;
#pragma unroll
    for (int r = 0; r < 8; ++r) {
      int v = vbase + wave * 16 + r + mo;
      float val = fmaxf(acc[r] + bb, 0.0f);
      if (cnt[v] == 0u) val = 0.0f;
      out[((long)batch * NHd + ch) * HWd + v] = val;   // [B,32,480,360]
    }
  }
}

// ---------------------------------------------------------------- launch
extern "C" void kernel_launch(void* const* d_in, const int* in_sizes, int n_in,
                              void* d_out, int out_size, void* d_ws, size_t ws_size,
                              hipStream_t stream) {
  const float* pt = (const float*)d_in[0];
  const int*   xy = (const int*)d_in[1];
  const float* s0 = (const float*)d_in[2];  const float* t0 = (const float*)d_in[3];
  const float* s1 = (const float*)d_in[4];  const float* t1 = (const float*)d_in[5];
  const float* s2 = (const float*)d_in[6];  const float* t2 = (const float*)d_in[7];
  const float* s3 = (const float*)d_in[8];  const float* t3 = (const float*)d_in[9];
  const float* W1 = (const float*)d_in[10]; const float* b1 = (const float*)d_in[11];
  const float* W2 = (const float*)d_in[12]; const float* b2 = (const float*)d_in[13];
  const float* W3 = (const float*)d_in[14]; const float* b3 = (const float*)d_in[15];
  const float* W4 = (const float*)d_in[16]; const float* b4 = (const float*)d_in[17];
  const float* Wc = (const float*)d_in[18]; const float* bc = (const float*)d_in[19];
  float* out = (float*)d_out;

  // workspace: enc[HW*512] u32 (reused per batch) + cnt[HW] + cnt2[HW]
  unsigned* enc  = (unsigned*)d_ws;
  unsigned* cnt  = enc + (size_t)HWd * C4d;
  unsigned* cnt2 = cnt + HWd;

  const size_t smem_mlp = (size_t)(64*32 + 128*64 + 256*128 + 16*256 + 8*2*16*256) * 2
                        + (size_t)(3*64 + 3*128 + 3*256 + 512) * 4
                        + 256 * 4;
  const size_t smem_cmp = (size_t)(32*512 + 8*16*512) * 2 + 32 * 4;

  hipFuncSetAttribute((const void*)mlp_scatter_kernel,
                      hipFuncAttributeMaxDynamicSharedMemorySize, (int)smem_mlp);
  hipFuncSetAttribute((const void*)compress_kernel,
                      hipFuncAttributeMaxDynamicSharedMemorySize, (int)smem_cmp);

  for (int b = 0; b < Bd; ++b) {
    zero_kernel<<<675, 256, 0, stream>>>(cnt, cnt2);
    count_kernel<<<Nd / 256, 256, 0, stream>>>(xy, cnt, b);
    init_kernel<<<4096, 256, 0, stream>>>(enc, cnt);
    mlp_scatter_kernel<<<Nd / 128, 256, smem_mlp, stream>>>(
        pt, xy, s0, t0, s1, t1, s2, t2, s3, t3,
        W1, b1, W2, b2, W3, b3, W4, b4, enc, cnt2, b);
    compress_kernel<<<HWd / 128, 256, smem_cmp, stream>>>(enc, cnt, Wc, bc, out, b);
  }
}
